// Block_79663053406569
// MI455X (gfx1250) — matrix operand order repaired and checked by reading, not compiled, not run on hardware
//
#include <hip/hip_runtime.h>
#include <hip/hip_bf16.h>

// ---------------------------------------------------------------------------
// Transformer block (B=128,T=256,C=384,H=6,D=64,F=1536) for gfx1250 (MI455X).
// Compute-bound (~130 GFLOP vs ~150MB traffic -> ~900 FLOP/B), so all GEMMs
// and both attention matmuls go through v_wmma_f32_16x16x32_bf16 (f32 accum).
// GEMM tiles are staged into LDS with global_load_async_to_lds_b128 (ASYNCcnt)
// when the builtin is available.
// ---------------------------------------------------------------------------

typedef __attribute__((ext_vector_type(16))) __bf16 v16bf;
typedef __attribute__((ext_vector_type(8)))  float  v8f;
typedef int i4v __attribute__((vector_size(16)));   // matches async-LDS builtin param

#define WMMA_BF16(Afrag, Bfrag, Cacc) \
  __builtin_amdgcn_wmma_f32_16x16x32_bf16(false, (Afrag), false, (Bfrag), (short)0, (Cacc), false, false)

#define AS1 __attribute__((address_space(1)))
#define AS3 __attribute__((address_space(3)))

#if defined(__has_builtin)
#if __has_builtin(__builtin_amdgcn_global_load_async_to_lds_b128) && \
    __has_builtin(__builtin_amdgcn_s_wait_asynccnt)
#define USE_ASYNC_LDS 1
#endif
#endif
#ifndef USE_ASYNC_LDS
#define USE_ASYNC_LDS 0
#endif

// A-fragment (16x32 bf16, row-major source): lane = m(0..15) + 16*kh.
// elements 0..7  -> K = kh*8 + e,  elements 8..15 -> K = 16 + kh*8 + e.
// caller passes rowk = &row[kh*8].
__device__ inline v16bf frag_a_bf16(const __bf16* rowk) {
  v16bf a;
#pragma unroll
  for (int e = 0; e < 8; ++e) { a[e] = rowk[e]; a[e + 8] = rowk[16 + e]; }
  return a;
}
// B-fragment (32x16, column-major load == rows of B^T): lane = n(0..15)+16*kh,
// elements 0..7 -> K = kh*8 + e, elements 8..15 -> K = 16 + kh*8 + e. caller passes rowk = &rowBt[kh*8].
__device__ inline v16bf frag_b_bf16(const __bf16* rowk) {
  v16bf b;
#pragma unroll
  for (int e = 0; e < 8; ++e) { b[e] = rowk[e]; b[e + 8] = rowk[16 + e]; }
  return b;
}

// 64 bytes (4 x b128) global -> LDS, per thread.  The instruction's immediate
// offset is added to BOTH the global and LDS addresses (ISA 08_async_tensor).
__device__ inline void stage64(const __bf16* g, __bf16* l) {
#if USE_ASYNC_LDS
  AS1 i4v* gp = (AS1 i4v*)g;
  AS3 i4v* lp = (AS3 i4v*)l;
  __builtin_amdgcn_global_load_async_to_lds_b128(gp, lp, 0, 0);
  __builtin_amdgcn_global_load_async_to_lds_b128(gp, lp, 16, 0);
  __builtin_amdgcn_global_load_async_to_lds_b128(gp, lp, 32, 0);
  __builtin_amdgcn_global_load_async_to_lds_b128(gp, lp, 48, 0);
#else
  const uint4* gs = reinterpret_cast<const uint4*>(g);
  uint4* ls = reinterpret_cast<uint4*>(l);
#pragma unroll
  for (int i = 0; i < 4; ++i) ls[i] = gs[i];
#endif
}

// ---------------------------------------------------------------------------
// Weight conversion: f32 -> bf16 with transpose.
// ---------------------------------------------------------------------------

// W[h][c][d] (6,384,64) -> Wt[(h*64+d)*384 + c]
__global__ void cvt_qkv_kernel(const float* __restrict__ w, __bf16* __restrict__ wt) {
  int i = blockIdx.x * 256 + threadIdx.x;
  if (i >= 6 * 384 * 64) return;
  int d = i & 63;
  int c = (i >> 6) % 384;
  int h = i / (384 * 64);
  wt[(size_t)(h * 64 + d) * 384 + c] = (__bf16)w[i];
}

// W[r][c] (R x Cc) -> Wt[c*R + r]  (i.e. [N][K] with N=Cc, K=R)
__global__ void cvt_t_kernel(const float* __restrict__ w, __bf16* __restrict__ wt, int R, int Cc) {
  int i = blockIdx.x * 256 + threadIdx.x;
  if (i >= R * Cc) return;
  int c = i % Cc, r = i / Cc;
  wt[(size_t)c * R + r] = (__bf16)w[i];
}

// ---------------------------------------------------------------------------
// LayerNorm: one wave32 per token (C=384 -> 12 elems/lane), bf16 output.
// ---------------------------------------------------------------------------
__global__ __launch_bounds__(256) void ln_kernel(const float* __restrict__ x,
                                                 const float* __restrict__ g,
                                                 const float* __restrict__ b,
                                                 __bf16* __restrict__ out,
                                                 int n_tokens) {
  int wave = threadIdx.x >> 5, lane = threadIdx.x & 31;
  int token = blockIdx.x * 8 + wave;
  if (token >= n_tokens) return;
  const float* row = x + (size_t)token * 384;
  float v[12], s = 0.f, s2 = 0.f;
#pragma unroll
  for (int i = 0; i < 12; ++i) {
    float t = row[lane + i * 32];
    v[i] = t; s += t; s2 += t * t;
  }
#pragma unroll
  for (int m = 16; m; m >>= 1) {
    s  += __shfl_xor(s,  m, 32);
    s2 += __shfl_xor(s2, m, 32);
  }
  float mu = s * (1.f / 384.f);
  float var = s2 * (1.f / 384.f) - mu * mu;
  float rinv = rsqrtf(var + 1e-5f);
  __bf16* orow = out + (size_t)token * 384;
#pragma unroll
  for (int i = 0; i < 12; ++i) {
    int c = lane + i * 32;
    orow[c] = (__bf16)((v[i] - mu) * rinv * g[c] + b[c]);
  }
}

// ---------------------------------------------------------------------------
// Tiled WMMA GEMM: Out[M,N] = A[M,K] * Bt[N,K]^T  (bf16 in, f32 accum).
// Block = 256 threads (8 waves), tile 128x128, K slabs of 64 through LDS
// (async-to-LDS staging).  Each wave owns a 32x64 sub-tile = 2x4 wmma tiles,
// 16 wmma issues per barrier round.
// MODE 0: bf16 out                  (Q, K projections)
// MODE 1: bf16 out scattered as V^T [B][H][D][T]   (V projection)
// MODE 2: f32 out = acc + bias[n] + res[m*N+n]     (O-proj, FFN2)
// MODE 3: bf16 out = relu(acc + bias[n])           (FFN1)
// ---------------------------------------------------------------------------
template <int MODE>
__global__ __launch_bounds__(256) void gemm_bf16_kernel(
    const __bf16* __restrict__ A, const __bf16* __restrict__ Bt,
    int M, int N, int K,
    const float* __restrict__ bias, const float* __restrict__ res,
    void* __restrict__ outp) {
  __shared__ __bf16 lA[128 * 64];   // 16KB
  __shared__ __bf16 lB[128 * 64];   // 16KB

  const int tid = threadIdx.x;
  const int m0 = blockIdx.y * 128;
  const int n0 = blockIdx.x * 128;
  const int lrow = tid >> 1;        // 0..127: row of A/Bt tile this thread loads
  const int lpart = tid & 1;        // which 32-elem (64B) half of the 64-wide slab
  const int wid = tid >> 5, lane = tid & 31;
  const int wm = (wid & 3) * 32;    // wave row offset inside 128-row tile
  const int wn = (wid >> 2) * 64;   // wave col offset inside 128-col tile
  const int l15 = lane & 15, kh = lane >> 4;

  v8f acc[2][4];
#pragma unroll
  for (int i = 0; i < 2; ++i)
#pragma unroll
    for (int j = 0; j < 4; ++j) acc[i][j] = {};

  const __bf16* gA = A + (size_t)(m0 + lrow) * K + lpart * 32;
  const __bf16* gB = Bt + (size_t)(n0 + lrow) * K + lpart * 32;
  __bf16* sA = &lA[lrow * 64 + lpart * 32];
  __bf16* sB = &lB[lrow * 64 + lpart * 32];

  for (int kk = 0; kk < K; kk += 64) {
    stage64(gA + kk, sA);
    stage64(gB + kk, sB);
    if (kk + 64 < K) {              // prefetch next slab -> global_prefetch
      __builtin_prefetch((const void*)(gA + kk + 64), 0, 0);
      __builtin_prefetch((const void*)(gB + kk + 64), 0, 0);
    }
#if USE_ASYNC_LDS
    __builtin_amdgcn_s_wait_asynccnt(0);
#endif
    __syncthreads();

#pragma unroll
    for (int k2 = 0; k2 < 64; k2 += 32) {
      v16bf bf[4];
#pragma unroll
      for (int ni = 0; ni < 4; ++ni)
        bf[ni] = frag_b_bf16(&lB[(wn + ni * 16 + l15) * 64 + k2 + kh * 8]);
#pragma unroll
      for (int mi = 0; mi < 2; ++mi) {
        v16bf af = frag_a_bf16(&lA[(wm + mi * 16 + l15) * 64 + k2 + kh * 8]);
#pragma unroll
        for (int ni = 0; ni < 4; ++ni)
          acc[mi][ni] = WMMA_BF16(af, bf[ni], acc[mi][ni]);
      }
    }
    __syncthreads();
  }

  // C/D fragment: element j of lane -> row (kh*8 + j), col l15 of the 16x16 tile
#pragma unroll
  for (int mi = 0; mi < 2; ++mi)
#pragma unroll
    for (int ni = 0; ni < 4; ++ni) {
      int gn = n0 + wn + ni * 16 + l15;
#pragma unroll
      for (int j = 0; j < 8; ++j) {
        int gm = m0 + wm + mi * 16 + kh * 8 + j;
        float va = acc[mi][ni][j];
        if constexpr (MODE == 0) {
          ((__bf16*)outp)[(size_t)gm * N + gn] = (__bf16)va;
        } else if constexpr (MODE == 1) {
          // gm = b*256 + t ; gn = h*64 + d ; vT[((b*6+h)*64+d)*256 + t]
          int bb = gm >> 8, t = gm & 255;
          int hh = gn >> 6, d = gn & 63;
          ((__bf16*)outp)[((((size_t)bb * 6 + hh) * 64 + d) << 8) + t] = (__bf16)va;
        } else if constexpr (MODE == 2) {
          ((float*)outp)[(size_t)gm * N + gn] = va + bias[gn] + res[(size_t)gm * N + gn];
        } else {
          float o = va + bias[gn];
          ((__bf16*)outp)[(size_t)gm * N + gn] = (__bf16)fmaxf(o, 0.f);
        }
      }
    }
}

// ---------------------------------------------------------------------------
// Causal attention, one workgroup per (b, h, 128-row Q block).  8 waves; each
// wave owns a 16-row strip.  S (f32, 128x256 = 128KB) lives in LDS (CDNA5
// WGPs have 320KB).  Q*K^T and P*V both use bf16 WMMA.  scale = 1/sqrt(64).
// q,k: [B*T][384] bf16 (head h at col h*64) ; vT: [B][H][64][256] bf16.
// ---------------------------------------------------------------------------
__global__ __launch_bounds__(256) void attn_kernel(const __bf16* __restrict__ q,
                                                   const __bf16* __restrict__ k,
                                                   const __bf16* __restrict__ vT,
                                                   __bf16* __restrict__ attn_out) {
  __shared__ float S[128 * 256];

  const int qblk = blockIdx.x & 1;
  const int bh = blockIdx.x >> 1;
  const int bb = bh / 6, h = bh % 6;
  const int wid = threadIdx.x >> 5, lane = threadIdx.x & 31;
  const int l15 = lane & 15, kh = lane >> 4;
  const int q0 = qblk * 128;
  const int row0 = wid * 16;                 // wave strip inside 128-row block
  const int qmax_w = q0 + row0 + 15;         // max query index in strip

  // ---- Phase 1: S = Q * K^T over all (valid) key tiles ----
  const __bf16* qbase = q + ((size_t)(bb * 256 + q0 + row0 + l15)) * 384 + h * 64;
  const int ntiles = (qmax_w >> 4) + 1;      // causal: skip fully-masked tiles
  for (int nt = 0; nt < ntiles; ++nt) {
    v8f acc = {};
#pragma unroll
    for (int kk = 0; kk < 64; kk += 32) {
      v16bf aq = frag_a_bf16(qbase + kk + kh * 8);
      v16bf bk = frag_b_bf16(k + ((size_t)(bb * 256 + nt * 16 + l15)) * 384 + h * 64 + kk + kh * 8);
      acc = WMMA_BF16(aq, bk, acc);
    }
#pragma unroll
    for (int j = 0; j < 8; ++j)
      S[(row0 + kh * 8 + j) * 256 + nt * 16 + l15] = acc[j];
  }
  __syncthreads();

  // ---- Phase 2: causal softmax, 2 lanes per row, in-place in S ----
  {
    const float scale = 0.125f;              // 1/sqrt(64)
    int r = row0 + (lane >> 1);
    int half = lane & 1;
    int qi = q0 + r;                         // valid keys: s in [0, qi]
    float* Srow = &S[r * 256];
    float mx = -3.0e38f;
    for (int s = half; s <= qi; s += 2) mx = fmaxf(mx, Srow[s]);
    mx = fmaxf(mx, __shfl_xor(mx, 1, 32));
    float mxs = mx * scale;
    float sum = 0.f;
    for (int s = half; s <= qi; s += 2) {
      float e = __expf(Srow[s] * scale - mxs);
      Srow[s] = e;
      sum += e;
    }
    sum += __shfl_xor(sum, 1, 32);
    float inv = 1.f / sum;
    for (int s = half; s <= qi; s += 2) Srow[s] *= inv;
    for (int s = qi + 1 + half; s < 256; s += 2) Srow[s] = 0.f;  // masked tail
  }
  __syncthreads();

  // ---- Phase 3: O = P * V  (P read from LDS f32 -> bf16 fragments) ----
  v8f co[4];
#pragma unroll
  for (int i = 0; i < 4; ++i) co[i] = {};
  const int schunks = (qmax_w >> 5) + 1;     // 32-wide K(=s) chunks
  for (int st = 0; st < schunks; ++st) {
    int s0 = st * 32;
    v16bf ap;
    {
      const float* p = &S[(row0 + l15) * 256 + s0];
#pragma unroll
      for (int e = 0; e < 8; ++e) {
        ap[e]     = (__bf16)p[kh * 8 + e];
        ap[e + 8] = (__bf16)p[16 + kh * 8 + e];
      }
    }
#pragma unroll
    for (int ni = 0; ni < 4; ++ni) {
      const __bf16* vp = vT + (((size_t)(bb * 6 + h) * 64 + ni * 16 + l15)) * 256 + s0 + kh * 8;
      v16bf bv = frag_b_bf16(vp);
      co[ni] = WMMA_BF16(ap, bv, co[ni]);
    }
  }
#pragma unroll
  for (int ni = 0; ni < 4; ++ni)
#pragma unroll
    for (int j = 0; j < 8; ++j) {
      int gm = bb * 256 + q0 + row0 + kh * 8 + j;
      int gn = h * 64 + ni * 16 + l15;
      attn_out[(size_t)gm * 384 + gn] = (__bf16)co[ni][j];
    }
}

// ---------------------------------------------------------------------------
// Host-side orchestration.
// ---------------------------------------------------------------------------
extern "C" void kernel_launch(void* const* d_in, const int* in_sizes, int n_in,
                              void* d_out, int out_size, void* d_ws, size_t ws_size,
                              hipStream_t stream) {
  (void)in_sizes; (void)n_in; (void)out_size; (void)ws_size;
  constexpr int Bb = 128, T = 256, C = 384, H = 6, F = 1536;
  constexpr size_t BT = (size_t)Bb * T;       // 32768 tokens

  const float* x     = (const float*)d_in[0];
  const float* ln1_g = (const float*)d_in[1];
  const float* ln1_b = (const float*)d_in[2];
  const float* Wq    = (const float*)d_in[3];
  const float* Wk    = (const float*)d_in[4];
  const float* Wv    = (const float*)d_in[5];
  const float* Wo    = (const float*)d_in[6];
  const float* bo    = (const float*)d_in[7];
  const float* ln2_g = (const float*)d_in[8];
  const float* ln2_b = (const float*)d_in[9];
  const float* W1    = (const float*)d_in[10];
  const float* b1    = (const float*)d_in[11];
  const float* W2    = (const float*)d_in[12];
  const float* b2    = (const float*)d_in[13];

  char* ws = (char*)d_ws;
  size_t o_h1 = 0;                       // bf16 [BT][C]
  size_t o_q  = o_h1 + BT * C * 2;       // bf16 [BT][C]
  size_t o_k  = o_q  + BT * C * 2;       // bf16 [BT][C]
  size_t o_v  = o_k  + BT * C * 2;       // bf16 [B][H][D][T] (transposed)
  size_t o_at = o_v  + BT * C * 2;       // bf16 [BT][C]
  size_t o_x2 = o_at + BT * C * 2;       // f32  [BT][C]
  size_t o_h2 = o_x2 + BT * C * 4;       // bf16 [BT][C]
  size_t o_wq = o_h2 + BT * C * 2;
  size_t o_wk = o_wq + (size_t)C * C * 2;
  size_t o_wv = o_wk + (size_t)C * C * 2;
  size_t o_wo = o_wv + (size_t)C * C * 2;
  size_t o_w1 = o_wo + (size_t)C * C * 2;
  size_t o_w2 = o_w1 + (size_t)C * F * 2;
  size_t o_f1 = o_q;                     // FFN hidden [BT][F] reuses q/k/v/attn (exact fit)

  __bf16* h1   = (__bf16*)(ws + o_h1);
  __bf16* qb   = (__bf16*)(ws + o_q);
  __bf16* kb   = (__bf16*)(ws + o_k);
  __bf16* vTb  = (__bf16*)(ws + o_v);
  __bf16* atb  = (__bf16*)(ws + o_at);
  float*  x2   = (float*)(ws + o_x2);
  __bf16* h2   = (__bf16*)(ws + o_h2);
  __bf16* wq_t = (__bf16*)(ws + o_wq);
  __bf16* wk_t = (__bf16*)(ws + o_wk);
  __bf16* wv_t = (__bf16*)(ws + o_wv);
  __bf16* wo_t = (__bf16*)(ws + o_wo);
  __bf16* w1_t = (__bf16*)(ws + o_w1);
  __bf16* w2_t = (__bf16*)(ws + o_w2);
  __bf16* f1   = (__bf16*)(ws + o_f1);

  // Weight conversion (f32 -> bf16, [N][K] layout)
  cvt_qkv_kernel<<<576, 256, 0, stream>>>(Wq, wq_t);
  cvt_qkv_kernel<<<576, 256, 0, stream>>>(Wk, wk_t);
  cvt_qkv_kernel<<<576, 256, 0, stream>>>(Wv, wv_t);
  cvt_t_kernel<<<(C * C + 255) / 256, 256, 0, stream>>>(Wo, wo_t, C, C);
  cvt_t_kernel<<<(C * F + 255) / 256, 256, 0, stream>>>(W1, w1_t, C, F);
  cvt_t_kernel<<<(F * C + 255) / 256, 256, 0, stream>>>(W2, w2_t, F, C);

  // LN1
  ln_kernel<<<(int)(BT / 8), 256, 0, stream>>>(x, ln1_g, ln1_b, h1, (int)BT);

  // Q, K, V projections (V scattered transposed)
  dim3 g384(C / 128, (int)(BT / 128));
  gemm_bf16_kernel<0><<<g384, 256, 0, stream>>>(h1, wq_t, (int)BT, C, C, nullptr, nullptr, qb);
  gemm_bf16_kernel<0><<<g384, 256, 0, stream>>>(h1, wk_t, (int)BT, C, C, nullptr, nullptr, kb);
  gemm_bf16_kernel<1><<<g384, 256, 0, stream>>>(h1, wv_t, (int)BT, C, C, nullptr, nullptr, vTb);

  // Causal attention
  attn_kernel<<<Bb * H * (T / 128), 256, 0, stream>>>(qb, kb, vTb, atb);

  // Output projection + residual -> x2 (f32)
  gemm_bf16_kernel<2><<<g384, 256, 0, stream>>>(atb, wo_t, (int)BT, C, C, bo, x, x2);

  // LN2
  ln_kernel<<<(int)(BT / 8), 256, 0, stream>>>(x2, ln2_g, ln2_b, h2, (int)BT);

  // FFN
  dim3 g1536(F / 128, (int)(BT / 128));
  gemm_bf16_kernel<3><<<g1536, 256, 0, stream>>>(h2, w1_t, (int)BT, F, C, b1, nullptr, f1);
  gemm_bf16_kernel<2><<<g384, 256, 0, stream>>>(f1, w2_t, (int)BT, C, F, b2, x2, (float*)d_out);
}
